// KANLinear_50294067036604
// MI455X (gfx1250) — compile-verified, matrix operand order checked by reading, not run on hardware
//
#include <hip/hip_runtime.h>

// ---------------------------------------------------------------------------
// KANLinear forward for MI455X (gfx1250, wave32, WMMA + async LDS copies).
//   out[b,o] = silu(x)[b,:] @ scale_base[o,:]  +  sum_{i,g} spline[b,i,g]*coeff[o,i,g]*ss
// Restructured as a single f16 GEMM: M=8192, N=1024, K=12288
//   A  [8192 x 12288] f16 : cols g*1024+i = spline basis g of in-dim i; cols 11264+i = silu(x)
//   Wt [1024 x 12288] f16 : N-major ("B transposed") so A/B fragments load identically
// f32 accumulation via v_wmma_f32_16x16x32_f16; staging via
// global_load_async_to_lds_b128 (ASYNCcnt) with double-buffered LDS.
// ---------------------------------------------------------------------------

typedef _Float16 v8h  __attribute__((ext_vector_type(8)));
typedef _Float16 v16h __attribute__((ext_vector_type(16)));
typedef float    v8f  __attribute__((ext_vector_type(8)));
typedef int      v4i  __attribute__((ext_vector_type(4)));

#define IN_DIM   1024
#define OUT_DIM  1024
#define TOKENS   8192
#define GK       11                         // GRID_SIZE + K
#define KD       (GK * IN_DIM + IN_DIM)     // 12288 total contraction length
#define NPTS     15                         // grid knots per in-dim

#define TM   128
#define TN   128
#define TK   32
#define PITCH 40                            // LDS row pitch in halves (pad vs bank conflicts)

#if __has_builtin(__builtin_amdgcn_global_load_async_to_lds_b128)
#define HAS_ASYNC_LDS 1
typedef __attribute__((address_space(1))) v4i gv4i_t;   // global int4
typedef __attribute__((address_space(3))) v4i sv4i_t;   // LDS int4
#else
#define HAS_ASYNC_LDS 0
#endif

__device__ __forceinline__ void async_cp16(const _Float16* g, _Float16* l) {
#if HAS_ASYNC_LDS
  // LDS operand: low 32 bits of a generic shared pointer == wave-relative LDS
  // byte offset; global operand: full 64-bit address.
  __builtin_amdgcn_global_load_async_to_lds_b128(
      (gv4i_t*)(unsigned long long)g,
      (sv4i_t*)(unsigned int)(unsigned long long)l,
      0, 0);
#else
  *(uint4*)l = *(const uint4*)g;            // fallback: synchronous staging
#endif
}

__device__ __forceinline__ void wait_async() {
#if HAS_ASYNC_LDS
#if __has_builtin(__builtin_amdgcn_s_wait_asynccnt)
  __builtin_amdgcn_s_wait_asynccnt(0);
#else
  asm volatile("s_wait_asynccnt 0x0" ::: "memory");
#endif
#endif
}

// ---------------------------------------------------------------------------
// Pass 1: build A = [spline bases | silu(x)] in f16. One thread per (b, i).
// ---------------------------------------------------------------------------
__global__ __launch_bounds__(256) void kan_build_A(
    const float* __restrict__ x, const float* __restrict__ grid,
    _Float16* __restrict__ A)
{
  const int tid = blockIdx.x * 256 + threadIdx.x;
  const int b = tid >> 10;          // / IN_DIM
  const int i = tid & (IN_DIM - 1);
  const float xv = x[(size_t)b * IN_DIM + i];

  float t[NPTS];
#pragma unroll
  for (int j = 0; j < NPTS; ++j) t[j] = grid[i * NPTS + j];

  float bs[NPTS - 1];                         // order-0 indicator bases
#pragma unroll
  for (int j = 0; j < NPTS - 1; ++j)
    bs[j] = (xv >= t[j] && xv < t[j + 1]) ? 1.0f : 0.0f;

#pragma unroll
  for (int kk = 1; kk <= 3; ++kk) {           // Cox-de Boor to order 3
#pragma unroll
    for (int j = 0; j < NPTS - 1 - kk; ++j) {
      const float l = (xv - t[j])          / (t[j + kk]     - t[j]);
      const float r = (t[j + kk + 1] - xv) / (t[j + kk + 1] - t[j + 1]);
      bs[j] = l * bs[j] + r * bs[j + 1];
    }
  }

  _Float16* dst = A + (size_t)b * KD;
#pragma unroll
  for (int g = 0; g < GK; ++g)
    dst[g * IN_DIM + i] = (_Float16)bs[g];    // coalesced across i

  const float s = xv / (1.0f + __expf(-xv));  // silu
  dst[GK * IN_DIM + i] = (_Float16)s;
}

// ---------------------------------------------------------------------------
// Pass 2: build Wt[o][k] (N-major weights) in f16. One thread per (o, i).
// ---------------------------------------------------------------------------
__global__ __launch_bounds__(256) void kan_build_W(
    const float* __restrict__ coeff, const float* __restrict__ scale_base,
    const float* __restrict__ scale_spline, _Float16* __restrict__ Wt)
{
  const int tid = blockIdx.x * 256 + threadIdx.x;
  const int o = tid >> 10;
  const int i = tid & (IN_DIM - 1);
  const float ss = scale_spline[0];
  const float* cp = coeff + ((size_t)o * IN_DIM + i) * GK;
  _Float16* dst = Wt + (size_t)o * KD;
#pragma unroll
  for (int g = 0; g < GK; ++g)
    dst[g * IN_DIM + i] = (_Float16)(cp[g] * ss);
  dst[GK * IN_DIM + i] = (_Float16)scale_base[(size_t)o * IN_DIM + i];
}

// ---------------------------------------------------------------------------
// Pass 3: GEMM  out[M,N] = A[M,K] * Wt[N,K]^T, f32 accumulation.
// 128x128 block tile, TK=32, double-buffered LDS filled by async copies.
// 8 waves; each wave owns 64(M) x 32(N) = 4x2 WMMA accumulators.
// ---------------------------------------------------------------------------
__global__ __launch_bounds__(256) void kan_gemm(
    const _Float16* __restrict__ A, const _Float16* __restrict__ Wt,
    float* __restrict__ out)
{
  __shared__ _Float16 lA[2][TM * PITCH];      // 2 x 128 x 32-tile, pitch 40
  __shared__ _Float16 lB[2][TN * PITCH];

  const int t    = threadIdx.x;
  const int lane = t & 31;
  const int wave = t >> 5;
  const int wm   = wave & 1;                  // 2 M-groups of 64 rows
  const int wn   = wave >> 1;                 // 4 N-groups of 32 cols
  const int bM   = blockIdx.y * TM;
  const int bN   = blockIdx.x * TN;

  // staging: each thread moves two 16B chunks of A and two of Wt per K-step
  const int rS = t >> 2;                      // row 0..63 (second chunk at +64)
  const int cS = (t & 3) * 8;                 // col offset in halves

  const _Float16* gA = A  + (size_t)(bM + rS) * KD + cS;
  const _Float16* gB = Wt + (size_t)(bN + rS) * KD + cS;

  v8f acc[4][2] = {};

  // per-lane fragment addressing (ISA 16-bit 16x32 layout):
  //   lane<16 : row lr, K = 0..7 & 16..23 ; lane>=16 : row lr, K = 8..15 & 24..31
  const int lr = lane & 15;
  const int kh = (lane >> 4) << 3;            // 0 or 8 halves

  auto stage = [&](int buf, int k0) {
    async_cp16(gA + k0,                   &lA[buf][rS * PITCH + cS]);
    async_cp16(gA + k0 + (size_t)64 * KD, &lA[buf][(rS + 64) * PITCH + cS]);
    async_cp16(gB + k0,                   &lB[buf][rS * PITCH + cS]);
    async_cp16(gB + k0 + (size_t)64 * KD, &lB[buf][(rS + 64) * PITCH + cS]);
  };

  auto compute = [&](int buf) {
    v16h af[4], bf[2];
#pragma unroll
    for (int f = 0; f < 4; ++f) {
      const _Float16* p = &lA[buf][(wm * 64 + f * 16 + lr) * PITCH + kh];
      const v8h lo = *(const v8h*)(p);        // K = kh .. kh+7
      const v8h hi = *(const v8h*)(p + 16);   // K = 16+kh .. 16+kh+7
      af[f] = __builtin_shufflevector(lo, hi, 0,1,2,3,4,5,6,7,8,9,10,11,12,13,14,15);
    }
#pragma unroll
    for (int g = 0; g < 2; ++g) {
      const _Float16* p = &lB[buf][(wn * 32 + g * 16 + lr) * PITCH + kh];
      const v8h lo = *(const v8h*)(p);
      const v8h hi = *(const v8h*)(p + 16);
      bf[g] = __builtin_shufflevector(lo, hi, 0,1,2,3,4,5,6,7,8,9,10,11,12,13,14,15);
    }
#pragma unroll
    for (int f = 0; f < 4; ++f)
#pragma unroll
      for (int g = 0; g < 2; ++g)
        acc[f][g] = __builtin_amdgcn_wmma_f32_16x16x32_f16(
            false, af[f], false, bf[g], (short)0, acc[f][g], false, false);
  };

  // prologue: fill buffer 0 for k=0
  stage(0, 0);
  wait_async();
  __syncthreads();

  // 2x-unrolled pipeline so buffer indices are compile-time constants.
  // KD / TK = 384 (even), so the loop is exact.
  for (int k0 = 0; k0 < KD; k0 += 2 * TK) {
    stage(1, k0 + TK);                        // prefetch next tile -> buf1
    compute(0);                               // consume buf0
    wait_async();
    __syncthreads();

    if (k0 + 2 * TK < KD)
      stage(0, k0 + 2 * TK);                  // prefetch -> buf0
    compute(1);                               // consume buf1
    wait_async();
    __syncthreads();
  }

  // epilogue: C/D layout — VGPR j holds row j (lanes 0-15) / 8+j (lanes 16-31)
  const int r0 = (lane >> 4) * 8;
#pragma unroll
  for (int f = 0; f < 4; ++f) {
#pragma unroll
    for (int g = 0; g < 2; ++g) {
      float* orow = out + (size_t)(bM + wm * 64 + f * 16 + r0) * OUT_DIM
                        + (bN + wn * 32 + g * 16 + lr);
#pragma unroll
      for (int j = 0; j < 8; ++j)
        orow[(size_t)j * OUT_DIM] = acc[f][g][j];
    }
  }
}

// ---------------------------------------------------------------------------
// Inputs (setup_inputs order): x, grid, coeff, scale_base, scale_spline (f32).
// Workspace: A (201.3 MB) + Wt (25.2 MB) f16.
// ---------------------------------------------------------------------------
extern "C" void kernel_launch(void* const* d_in, const int* in_sizes, int n_in,
                              void* d_out, int out_size, void* d_ws, size_t ws_size,
                              hipStream_t stream) {
  const float* x   = (const float*)d_in[0];
  const float* gr  = (const float*)d_in[1];
  const float* co  = (const float*)d_in[2];
  const float* sb  = (const float*)d_in[3];
  const float* ssp = (const float*)d_in[4];
  float* out = (float*)d_out;

  _Float16* A  = (_Float16*)d_ws;
  _Float16* Wt = (_Float16*)((char*)d_ws + (size_t)TOKENS * KD * sizeof(_Float16));

  kan_build_A<<<(TOKENS * IN_DIM) / 256, 256, 0, stream>>>(x, gr, A);
  kan_build_W<<<(OUT_DIM * IN_DIM) / 256, 256, 0, stream>>>(co, sb, ssp, Wt);
  kan_gemm<<<dim3(OUT_DIM / TN, TOKENS / TM), 256, 0, stream>>>(A, Wt, out);
}
